// MultiHeadedLocalAttention_48069273977134
// MI455X (gfx1250) — matrix operand order repaired and checked by reading, not compiled, not run on hardware
//
#include <hip/hip_runtime.h>
#include <stdint.h>

// ---------------------------------------------------------------------------
// Problem constants
// ---------------------------------------------------------------------------
#define TOK     100352      // 32*56*56 tokens
#define DIMD    384
#define QKVN    1152        // 3*inner
#define NHEAD   12
#define DH      32
#define WW      64          // tokens per 8x8 window
#define NWB     49          // windows per batch image (7*7)
#define NWIN    1568        // 32*49 windows total
#define LDA     40          // padded LDS row stride (ushorts), keeps 16B align

#ifndef __has_builtin
#define __has_builtin(x) 0
#endif

#if __has_builtin(__builtin_amdgcn_tensor_load_to_lds) && \
    __has_builtin(__builtin_amdgcn_s_wait_tensorcnt)
#define HAVE_TDM 1
#else
#define HAVE_TDM 0
#endif

// ---------------------------------------------------------------------------
// Vector / WMMA types (plain clang ext vectors: POD, union-safe)
// ---------------------------------------------------------------------------
typedef __attribute__((ext_vector_type(16))) __bf16         v16bf;
typedef __attribute__((ext_vector_type(8)))  float          v8f;
typedef __attribute__((ext_vector_type(4)))  unsigned int   u32x4;
typedef __attribute__((ext_vector_type(4)))  unsigned short u16x4;
typedef __attribute__((ext_vector_type(4)))  float          f32x4;
typedef __attribute__((ext_vector_type(8)))  int            i32x8;
typedef __attribute__((ext_vector_type(4)))  int            i32x4;

union Frag {            // 16 bf16 values = 8 VGPRs = one WMMA A/B operand
    u32x4 q[2];
    v16bf bf;
};

__device__ __forceinline__ unsigned short f2bf(float f) {
    unsigned int u = __float_as_uint(f);
    unsigned int r = u + 0x7FFFu + ((u >> 16) & 1u);   // round-nearest-even
    return (unsigned short)(r >> 16);
}

__device__ __forceinline__ v8f wmma_bf16(const Frag& a, const Frag& b, v8f c) {
    return __builtin_amdgcn_wmma_f32_16x16x32_bf16(
        /*neg_a=*/false, a.bf, /*neg_b=*/false, b.bf,
        /*c_mod=*/(short)0, c, /*reuse_a=*/false, /*reuse_b=*/false);
}

// ---------------------------------------------------------------------------
// Tensor Data Mover: DMA a 2D bf16 tile (tile_d0 x tile_d1 elements) from a
// row-major tensor (row length tensor_d0, tensor_d1 rows, stride in elems)
// into LDS at lds_byte_off. Pad fields replicate the LDA=40 padded layout:
// after every 16 DWORDs (one 64B row) insert 4 DWORDs (16B) -> 80B/row.
// ---------------------------------------------------------------------------
__device__ __forceinline__ void tdm_load_tile_bf16(
    unsigned lds_byte_off, const void* gaddr,
    unsigned tensor_d0, unsigned tensor_d1, unsigned stride_elems,
    unsigned tile_d0, unsigned tile_d1) {
#if HAVE_TDM
    unsigned long long ga = (unsigned long long)(uintptr_t)gaddr;
    u32x4 g0;
    g0.x = 0x1u;                                          // count=1, user D#
    g0.y = lds_byte_off;                                  // lds_addr (bytes)
    g0.z = (unsigned)(ga & 0xffffffffu);                  // global_addr lo
    g0.w = (unsigned)((ga >> 32) & 0x01ffffffu)           // global_addr hi
         | 0x80000000u;                                   // type=2 ("image")
    i32x8 g1;
    // data_size=1 (2B) | pad_enable | pad_interval=3 (16 DW) | pad_amount=3 (4 DW)
    g1[0] = (int)((1u << 16) | (1u << 20) | (3u << 22) | (3u << 25));
    g1[1] = (int)((tensor_d0 & 0xffffu) << 16);           // tensor_dim0 lo16
    g1[2] = (int)(((tensor_d0 >> 16) & 0xffffu) |         // tensor_dim0 hi16
                  ((tensor_d1 & 0xffffu) << 16));         // tensor_dim1 lo16
    g1[3] = (int)(((tensor_d1 >> 16) & 0xffffu) |         // tensor_dim1 hi16
                  ((tile_d0 & 0xffffu) << 16));           // tile_dim0
    g1[4] = (int)(tile_d1 & 0xffffu);                     // tile_dim1 (dim2=0)
    g1[5] = (int)stride_elems;                            // tensor_dim0_stride lo
    g1[6] = 0;
    g1[7] = 0;
    i32x4 z4 = {0, 0, 0, 0};
#if defined(__clang_major__) && (__clang_major__ >= 23)
    i32x8 z8 = {0, 0, 0, 0, 0, 0, 0, 0};
    __builtin_amdgcn_tensor_load_to_lds(g0, g1, z4, z4, z8, 0);
#else
    __builtin_amdgcn_tensor_load_to_lds(g0, g1, z4, z4, 0);
#endif
#endif
}

// ---------------------------------------------------------------------------
// Kernel 0: convert both weight matrices fp32 -> bf16 (1.2 MB, one shot)
// ---------------------------------------------------------------------------
__global__ void cvt_weights(const float* __restrict__ Wq, const float* __restrict__ Wo,
                            unsigned short* __restrict__ oq, unsigned short* __restrict__ oo) {
    int i = blockIdx.x * 256 + threadIdx.x;
    if (i < DIMD * QKVN) oq[i] = f2bf(Wq[i]);
    if (i < DIMD * DIMD) oo[i] = f2bf(Wo[i]);
}

// ---------------------------------------------------------------------------
// Kernel 1: QKV GEMM  [100352,384](f32) x [384,1152](bf16) + bias
// 128x128 block tile, 8 waves, each wave 64x32 (4x2 WMMA tiles), K-step 32.
// Staging is split load-phase / store-phase so the 4 global loads issue as
// one clause with a single wait instead of 4 serialized load->wait pairs.
// Epilogue fuses bias add + window partition: scatter into q/k/v buffers
// laid out [window, head, 64 tokens, 32 dims] in bf16.
// ---------------------------------------------------------------------------
__global__ __launch_bounds__(256) void qkv_gemm(
    const float* __restrict__ X, const unsigned short* __restrict__ Wb,
    const float* __restrict__ bias,
    unsigned short* __restrict__ Q, unsigned short* __restrict__ Kd,
    unsigned short* __restrict__ V) {
    __shared__ __align__(16) unsigned short As[128 * LDA];   // [row][k] bf16
    __shared__ __align__(16) unsigned short Bs[128 * LDA];   // [n][k]  bf16 (transposed)

    const int tid  = threadIdx.x;
    const int bm   = blockIdx.x;          // 0..783
    const int bn   = blockIdx.y;          // 0..8
    const int lane = tid & 31;
    const int w    = tid >> 5;
    const int wm   = w >> 2;              // 0..1  (64-row half)
    const int wn   = w & 3;               // 0..3  (32-col quarter)
    const int lm   = lane & 15;
    const int hi   = lane >> 4;

    const v8f vzero = {0.f,0.f,0.f,0.f,0.f,0.f,0.f,0.f};
    v8f acc[4][2];
#pragma unroll
    for (int i = 0; i < 4; ++i)
#pragma unroll
        for (int j = 0; j < 2; ++j) acc[i][j] = vzero;

    const int ar = tid >> 3;              // A-stage row base   0..31
    const int ac = (tid & 7) * 4;         // A-stage col        0..28
    const int bk = tid >> 5;              // B-stage k base     0..7
    const int bnn = (tid & 31) * 4;       // B-stage n base     0..124

    for (int k0 = 0; k0 < DIMD; k0 += 32) {
        // --- A tile: 128x32 fp32 -> bf16 LDS. Load phase first (one clause) ---
        f32x4 fa[4];
#pragma unroll
        for (int p = 0; p < 4; ++p) {
            const float* gp = X + (size_t)(bm * 128 + ar + p * 32) * DIMD + k0 + ac;
            if (k0 + 32 < DIMD) __builtin_prefetch(gp + 32, 0, 3);
            fa[p] = *(const f32x4*)gp;
        }
        // --- B tile: W[k0..+32, bn*128..+128] bf16, load phase ---
        u16x4 wv[4];
#pragma unroll
        for (int p = 0; p < 4; ++p) {
            const unsigned short* gp =
                Wb + (size_t)(k0 + bk + p * 8) * QKVN + bn * 128 + bnn;
            if (k0 + 32 < DIMD) __builtin_prefetch(gp + 32 * QKVN, 0, 3);
            wv[p] = *(const u16x4*)gp;
        }
        // --- convert/store phase ---
#pragma unroll
        for (int p = 0; p < 4; ++p) {
            u16x4 h;
            h.x = f2bf(fa[p].x); h.y = f2bf(fa[p].y);
            h.z = f2bf(fa[p].z); h.w = f2bf(fa[p].w);
            *(u16x4*)&As[(ar + p * 32) * LDA + ac] = h;
        }
#pragma unroll
        for (int p = 0; p < 4; ++p) {
            const int kk = bk + p * 8;
            Bs[(bnn + 0) * LDA + kk] = wv[p].x;
            Bs[(bnn + 1) * LDA + kk] = wv[p].y;
            Bs[(bnn + 2) * LDA + kk] = wv[p].z;
            Bs[(bnn + 3) * LDA + kk] = wv[p].w;
        }
        __syncthreads();

        // --- fragments + 8 WMMAs ---
        Frag a[4], b[2];
#pragma unroll
        for (int i = 0; i < 4; ++i) {
            // A layout: lane(row m), elems K = (j<8? j : j+8) + 8*hi
            const unsigned short* p = &As[(wm * 64 + i * 16 + lm) * LDA + 8 * hi];
            a[i].q[0] = *(const u32x4*)p;
            a[i].q[1] = *(const u32x4*)(p + 16);
        }
#pragma unroll
        for (int j = 0; j < 2; ++j) {
            // B layout: lane(col n), elems K = (j<8? j : j+8) + 8*hi
            const unsigned short* p = &Bs[(wn * 32 + j * 16 + lm) * LDA + 8 * hi];
            b[j].q[0] = *(const u32x4*)p;
            b[j].q[1] = *(const u32x4*)(p + 16);
        }
#pragma unroll
        for (int i = 0; i < 4; ++i)
#pragma unroll
            for (int j = 0; j < 2; ++j)
                acc[i][j] = wmma_bf16(a[i], b[j], acc[i][j]);
        __syncthreads();
    }

    // --- epilogue: bias + window-partition scatter (bf16) ---
#pragma unroll
    for (int j = 0; j < 2; ++j) {
        const int c    = bn * 128 + wn * 32 + j * 16 + lm;   // 0..1151
        const int sel  = (c >= 768) ? 2 : (c >= 384 ? 1 : 0);
        const int ic   = c - sel * DIMD;
        const int head = ic >> 5;
        const int d    = ic & 31;
        const float bb = bias[c];
        unsigned short* dst = sel == 0 ? Q : (sel == 1 ? Kd : V);
#pragma unroll
        for (int i = 0; i < 4; ++i) {
#pragma unroll
            for (int r = 0; r < 8; ++r) {
                const int t   = bm * 128 + wm * 64 + i * 16 + hi * 8 + r;  // token id
                const int b_  = t / 3136;
                const int rem = t - b_ * 3136;
                const int y   = rem / 56;
                const int x   = rem - y * 56;
                const int widx = b_ * NWB + (y >> 3) * 7 + (x >> 3);
                const int ti   = (y & 7) * 8 + (x & 7);
                dst[((size_t)(widx * NHEAD + head) * WW + ti) * DH + d] =
                    f2bf(acc[i][j][r] + bb);
            }
        }
    }
}

// ---------------------------------------------------------------------------
// Kernel 2: fused window attention. One wave per (window, head).
// S = Q K^T (16 WMMAs), softmax in registers (shfl_xor row reductions),
// P staged to LDS (D-layout -> A-layout), O = P V (16 WMMAs),
// epilogue merges windows back to token-major bf16 [TOK, 384].
// ---------------------------------------------------------------------------
__global__ __launch_bounds__(32) void attn_kernel(
    const unsigned short* __restrict__ Q, const unsigned short* __restrict__ Kd,
    const unsigned short* __restrict__ V, unsigned short* __restrict__ AO) {
    __shared__ __align__(16) unsigned short P[64 * 64];   // softmax probs bf16
    __shared__ __align__(16) unsigned short Vt[32 * 64];  // V transposed [dh][tok]
    __shared__ __align__(16) unsigned short Ot[64 * 32];  // output staging

    const int wh   = blockIdx.x;              // 0..18815
    const int widx = wh / NHEAD;
    const int head = wh - widx * NHEAD;
    const unsigned short* q = Q  + (size_t)wh * (WW * DH);
    const unsigned short* k = Kd + (size_t)wh * (WW * DH);
    const unsigned short* v = V  + (size_t)wh * (WW * DH);

    const int lane = threadIdx.x;
    const int lm   = lane & 15;
    const int hi   = lane >> 4;

    // transpose V -> Vt[dh][token]: load both rows first (one clause), then scatter
    {
        u16x4 vr[2][8];
#pragma unroll
        for (int rr = 0; rr < 2; ++rr) {
            const unsigned short* src = v + (lane * 2 + rr) * DH;
#pragma unroll
            for (int c = 0; c < 8; ++c) vr[rr][c] = *(const u16x4*)(src + c * 4);
        }
#pragma unroll
        for (int rr = 0; rr < 2; ++rr) {
            const int row = lane * 2 + rr;
#pragma unroll
            for (int c = 0; c < 8; ++c) {
                Vt[(c * 4 + 0) * 64 + row] = vr[rr][c].x;
                Vt[(c * 4 + 1) * 64 + row] = vr[rr][c].y;
                Vt[(c * 4 + 2) * 64 + row] = vr[rr][c].z;
                Vt[(c * 4 + 3) * 64 + row] = vr[rr][c].w;
            }
        }
    }
    __syncthreads();

    // ---- S = Q K^T : 4x4 tiles of 16x16, K(contraction)=32 ----
    Frag a[4], b[4];
#pragma unroll
    for (int i = 0; i < 4; ++i) {
        const unsigned short* p = q + (i * 16 + lm) * DH + 8 * hi;
        a[i].q[0] = *(const u32x4*)p;
        a[i].q[1] = *(const u32x4*)(p + 16);
    }
#pragma unroll
    for (int j = 0; j < 4; ++j) {
        const unsigned short* p = k + (j * 16 + lm) * DH + 8 * hi;
        b[j].q[0] = *(const u32x4*)p;
        b[j].q[1] = *(const u32x4*)(p + 16);
    }
    const v8f vzero = {0.f,0.f,0.f,0.f,0.f,0.f,0.f,0.f};
    v8f s[4][4];
#pragma unroll
    for (int i = 0; i < 4; ++i)
#pragma unroll
        for (int j = 0; j < 4; ++j) {
            s[i][j] = vzero;
            s[i][j] = wmma_bf16(a[i], b[j], s[i][j]);
        }

    // ---- softmax over rows (row elements live across the 16-lane half) ----
    const float scale = 0.17677669529663687f;   // 1/sqrt(32)
#pragma unroll
    for (int i = 0; i < 4; ++i) {
        float mx[8];
#pragma unroll
        for (int r = 0; r < 8; ++r) mx[r] = -3.0e38f;
#pragma unroll
        for (int j = 0; j < 4; ++j)
#pragma unroll
            for (int r = 0; r < 8; ++r) mx[r] = fmaxf(mx[r], s[i][j][r]);
#pragma unroll
        for (int st = 1; st < 16; st <<= 1)
#pragma unroll
            for (int r = 0; r < 8; ++r)
                mx[r] = fmaxf(mx[r], __shfl_xor(mx[r], st, 32));

        float sm[8];
        float e[4][8];
#pragma unroll
        for (int r = 0; r < 8; ++r) sm[r] = 0.f;
#pragma unroll
        for (int j = 0; j < 4; ++j)
#pragma unroll
            for (int r = 0; r < 8; ++r) {
                e[j][r] = __expf((s[i][j][r] - mx[r]) * scale);
                sm[r] += e[j][r];
            }
#pragma unroll
        for (int st = 1; st < 16; st <<= 1)
#pragma unroll
            for (int r = 0; r < 8; ++r) sm[r] += __shfl_xor(sm[r], st, 32);

#pragma unroll
        for (int r = 0; r < 8; ++r) {
            const float inv = 1.0f / sm[r];
            const int row = i * 16 + hi * 8 + r;
#pragma unroll
            for (int j = 0; j < 4; ++j)
                P[row * 64 + j * 16 + lm] = f2bf(e[j][r] * inv);
        }
    }
    __syncthreads();

    // ---- O = P V : 4x2 tiles, contraction K=64 in two 32-steps ----
    v8f o[4][2];
#pragma unroll
    for (int i = 0; i < 4; ++i)
#pragma unroll
        for (int j = 0; j < 2; ++j) o[i][j] = vzero;

#pragma unroll
    for (int kk = 0; kk < 2; ++kk) {
        Frag bv[2];
#pragma unroll
        for (int j = 0; j < 2; ++j) {
            const unsigned short* p = &Vt[(j * 16 + lm) * 64 + kk * 32 + 8 * hi];
            bv[j].q[0] = *(const u32x4*)p;
            bv[j].q[1] = *(const u32x4*)(p + 16);
        }
#pragma unroll
        for (int i = 0; i < 4; ++i) {
            Frag ap;
            const unsigned short* p = &P[(i * 16 + lm) * 64 + kk * 32 + 8 * hi];
            ap.q[0] = *(const u32x4*)p;
            ap.q[1] = *(const u32x4*)(p + 16);
#pragma unroll
            for (int j = 0; j < 2; ++j) o[i][j] = wmma_bf16(ap, bv[j], o[i][j]);
        }
    }

    // stage result to LDS (bf16), then coalesced merged-window store
#pragma unroll
    for (int i = 0; i < 4; ++i)
#pragma unroll
        for (int j = 0; j < 2; ++j)
#pragma unroll
            for (int r = 0; r < 8; ++r)
                Ot[(i * 16 + hi * 8 + r) * 32 + j * 16 + lm] = f2bf(o[i][j][r]);
    __syncthreads();

    const int b_ = widx / NWB;
    const int wr = widx - b_ * NWB;
    const int wy = wr / 7;
    const int wx = wr - wy * 7;
#pragma unroll
    for (int rr = 0; rr < 2; ++rr) {
        const int row = lane * 2 + rr;
        const int py = row >> 3, px = row & 7;
        const size_t t = (size_t)b_ * 3136 + (size_t)(wy * 8 + py) * 56 + (wx * 8 + px);
        unsigned short* dst = AO + t * DIMD + head * DH;
        const unsigned short* src = &Ot[row * 32];
#pragma unroll
        for (int d = 0; d < 32; d += 8)
            *(u32x4*)(dst + d) = *(const u32x4*)(src + d);
    }
}

// ---------------------------------------------------------------------------
// Kernel 3: output GEMM  [100352,384](bf16) x [384,384](bf16) + bias -> fp32
// A tile staged via the Tensor Data Mover (async DMA, TENSORcnt) with the
// D# pad fields reproducing the LDA=40 padded LDS layout.
// ---------------------------------------------------------------------------
__global__ __launch_bounds__(256) void out_gemm(
    const unsigned short* __restrict__ A, const unsigned short* __restrict__ Wb,
    const float* __restrict__ bias, float* __restrict__ out) {
    __shared__ __align__(16) unsigned short As[128 * LDA];
    __shared__ __align__(16) unsigned short Bs[128 * LDA];

    const int tid  = threadIdx.x;
    const int bm   = blockIdx.x;          // 0..783
    const int bn   = blockIdx.y;          // 0..2
    const int lane = tid & 31;
    const int w    = tid >> 5;
    const int wm   = w >> 2;
    const int wn   = w & 3;
    const int lm   = lane & 15;
    const int hi   = lane >> 4;

    const v8f vzero = {0.f,0.f,0.f,0.f,0.f,0.f,0.f,0.f};
    v8f acc[4][2];
#pragma unroll
    for (int i = 0; i < 4; ++i)
#pragma unroll
        for (int j = 0; j < 2; ++j) acc[i][j] = vzero;

    const int bk  = tid >> 5;             // B-stage k base 0..7
    const int bnn = (tid & 31) * 4;       // B-stage n base 0..124

    for (int k0 = 0; k0 < DIMD; k0 += 32) {
#if HAVE_TDM
        // --- A tile (128 rows x 32 cols bf16) via TDM async DMA, wave 0 only ---
        if (tid < 32) {
            tdm_load_tile_bf16((unsigned)(uintptr_t)As,
                               A + (size_t)(bm * 128) * DIMD + k0,
                               /*tensor_d0=*/DIMD, /*tensor_d1=*/TOK,
                               /*stride=*/DIMD, /*tile_d0=*/32, /*tile_d1=*/128);
        }
#else
        {   // A already bf16: copy 128x32 tile
            const int r = tid >> 2;           // 0..63
            const int c = (tid & 3) * 8;      // 0..24
            u32x4 av[2];
#pragma unroll
            for (int p = 0; p < 2; ++p)
                av[p] = *(const u32x4*)(A + (size_t)(bm * 128 + r + p * 64) * DIMD + k0 + c);
#pragma unroll
            for (int p = 0; p < 2; ++p)
                *(u32x4*)&As[(r + p * 64) * LDA + c] = av[p];
        }
#endif
        {   // B bf16, store transposed Bs[n][k]: load phase then scatter phase
            u16x4 wv[4];
#pragma unroll
            for (int p = 0; p < 4; ++p)
                wv[p] = *(const u16x4*)(Wb + (size_t)(k0 + bk + p * 8) * DIMD +
                                        bn * 128 + bnn);
#pragma unroll
            for (int p = 0; p < 4; ++p) {
                const int kk = bk + p * 8;
                Bs[(bnn + 0) * LDA + kk] = wv[p].x;
                Bs[(bnn + 1) * LDA + kk] = wv[p].y;
                Bs[(bnn + 2) * LDA + kk] = wv[p].z;
                Bs[(bnn + 3) * LDA + kk] = wv[p].w;
            }
        }
#if HAVE_TDM
        if (tid < 32) __builtin_amdgcn_s_wait_tensorcnt(0);   // DMA complete
#endif
        __syncthreads();

        Frag a[4], b[2];
#pragma unroll
        for (int i = 0; i < 4; ++i) {
            const unsigned short* p = &As[(wm * 64 + i * 16 + lm) * LDA + 8 * hi];
            a[i].q[0] = *(const u32x4*)p;
            a[i].q[1] = *(const u32x4*)(p + 16);
        }
#pragma unroll
        for (int j = 0; j < 2; ++j) {
            const unsigned short* p = &Bs[(wn * 32 + j * 16 + lm) * LDA + 8 * hi];
            b[j].q[0] = *(const u32x4*)p;
            b[j].q[1] = *(const u32x4*)(p + 16);
        }
#pragma unroll
        for (int i = 0; i < 4; ++i)
#pragma unroll
            for (int j = 0; j < 2; ++j)
                acc[i][j] = wmma_bf16(a[i], b[j], acc[i][j]);
        __syncthreads();
    }

#pragma unroll
    for (int j = 0; j < 2; ++j) {
        const int c = bn * 128 + wn * 32 + j * 16 + lm;
        const float bb = bias[c];
#pragma unroll
        for (int i = 0; i < 4; ++i)
#pragma unroll
            for (int r = 0; r < 8; ++r) {
                const int t = bm * 128 + wm * 64 + i * 16 + hi * 8 + r;
                out[(size_t)t * DIMD + c] = acc[i][j][r] + bb;
            }
    }
}

// ---------------------------------------------------------------------------
// Host-side launch
// ---------------------------------------------------------------------------
extern "C" void kernel_launch(void* const* d_in, const int* in_sizes, int n_in,
                              void* d_out, int out_size, void* d_ws, size_t ws_size,
                              hipStream_t stream) {
    const float* x    = (const float*)d_in[0];
    const float* Wqkv = (const float*)d_in[1];
    const float* bqkv = (const float*)d_in[2];
    const float* Wout = (const float*)d_in[3];
    const float* bout = (const float*)d_in[4];
    float* out = (float*)d_out;

    char* ws = (char*)d_ws;
    unsigned short* wqkv_b = (unsigned short*)ws;                       // 442368 elems
    unsigned short* wout_b = (unsigned short*)(ws + 884736);            // 147456 elems
    unsigned short* q_ws   = (unsigned short*)(ws + 1179648);           // 38.5M bf16
    unsigned short* k_ws   = q_ws + (size_t)NWIN * NHEAD * WW * DH;
    unsigned short* v_ws   = k_ws + (size_t)NWIN * NHEAD * WW * DH;
    unsigned short* a_ws   = v_ws + (size_t)NWIN * NHEAD * WW * DH;     // [TOK,384] bf16

    cvt_weights<<<1728, 256, 0, stream>>>(Wqkv, Wout, wqkv_b, wout_b);
    qkv_gemm<<<dim3(784, 9), 256, 0, stream>>>(x, wqkv_b, bqkv, q_ws, k_ws, v_ws);
    attn_kernel<<<NWIN * NHEAD, 32, 0, stream>>>(q_ws, k_ws, v_ws, a_ws);
    out_gemm<<<dim3(784, 3), 256, 0, stream>>>(a_ws, wout_b, bout, out);
}
